// BertEmbedding_91173565760191
// MI455X (gfx1250) — matrix operand order repaired and checked by reading, not compiled, not run on hardware
//
#include <hip/hip_runtime.h>

typedef unsigned short u16;
typedef unsigned int   u32;
typedef __attribute__((ext_vector_type(8)))  __bf16 v8bf;
typedef __attribute__((ext_vector_type(16))) __bf16 v16bf;
typedef __attribute__((ext_vector_type(8)))  float  v8f;

#define B_   64
#define S_   512
#define H_   768
#define E_   512
#define T_   256
#define ROWS 32          // output rows (t's) per block
#define LDA  776         // padded LDS row stride in bf16 elements (768 + 8 -> bank-conflict-free)

union V16u { v16bf v; v8bf h[2]; };

__device__ __forceinline__ int lower_bound_dev(const int* __restrict__ a, int n, int key) {
  int lo = 0, hi = n;
  while (lo < hi) { int mid = (lo + hi) >> 1; if (a[mid] < key) lo = mid + 1; else hi = mid; }
  return lo;
}

// exact split: v == bf16(hi) + bf16(lo) up to ~2^-17 relative
__device__ __forceinline__ void split_bf16(float v, u16& hi, u16& lo) {
  u32 u = __float_as_uint(v);
  hi = (u16)(u >> 16);
  float r = v - __uint_as_float(u & 0xFFFF0000u);   // exact (Sterbenz)
  lo = (u16)(__float_as_uint(r) >> 16);
}

// ---------------- W split kernel: fp32 -> (Whi, Wlo) bf16 ----------------
__global__ __launch_bounds__(256) void split_w_kernel(const float* __restrict__ W,
                                                      u16* __restrict__ Whi,
                                                      u16* __restrict__ Wlo) {
  int i = blockIdx.x * 256 + threadIdx.x;
  if (i < E_ * H_) {
    u16 h, l;
    split_bf16(W[i], h, l);
    Whi[i] = h;
    Wlo[i] = l;
  }
}

// ---------- fused kernel: segment-mean pool (LDS) + bf16x3 WMMA GEMM ----------
__global__ __launch_bounds__(256) void fused_pool_gemm(
    const float* __restrict__ hs,   // [B,S,H]
    const int*   __restrict__ wid,  // [B,S] sorted
    const int*   __restrict__ tlen, // [B]
    const u16*   __restrict__ Whi,  // [E,H] bf16 hi
    const u16*   __restrict__ Wlo,  // [E,H] bf16 lo
    const float* __restrict__ bias, // [E]
    float*       __restrict__ out)  // [B,T,E]
{
  extern __shared__ u16 smem[];
  u16* Ahi = smem;                         // [ROWS][LDA]
  u16* Alo = smem + ROWS * LDA;            // [ROWS][LDA]
  int* sb  = (int*)(smem + 2 * ROWS * LDA);// [ROWS+1] segment boundaries

  const int blk  = blockIdx.x;
  const int b    = blk >> 3;               // 8 t-tiles of 32 per batch
  const int t0   = (blk & 7) * ROWS;
  const int tid  = threadIdx.x;
  const int lane = tid & 31;
  const int wv   = tid >> 5;               // wave id 0..7
  const int tl   = tlen[b];
  const int* wrow = wid + b * S_;

  if (tid <= ROWS) sb[tid] = lower_bound_dev(wrow, S_, t0 + tid);
  __syncthreads();

  // ---- phase 1: pooled rows -> LDS as split bf16 (hi, lo) ----
  for (int rr = 0; rr < 4; ++rr) {
    const int r  = wv + rr * 8;            // each wave owns 4 of the 32 rows
    const int t  = t0 + r;
    const int s0 = sb[r], s1 = sb[r + 1];
    const int cnt = s1 - s0;
    const float inv = (cnt > 0 && t < tl) ? (1.0f / (float)cnt) : 0.0f;

    float4 acc[6];
    #pragma unroll
    for (int j = 0; j < 6; ++j) acc[j] = make_float4(0.f, 0.f, 0.f, 0.f);

    if (inv != 0.0f) {
      for (int s = s0; s < s1; ++s) {
        const float4* row4 = (const float4*)(hs + (size_t)(b * S_ + s) * H_);
        #pragma unroll
        for (int j = 0; j < 6; ++j) {       // 32 lanes * 6 float4 = 768 floats
          float4 v = row4[lane + 32 * j];
          acc[j].x += v.x; acc[j].y += v.y; acc[j].z += v.z; acc[j].w += v.w;
        }
      }
    }

    #pragma unroll
    for (int j = 0; j < 6; ++j) {
      const int h = (lane + 32 * j) * 4;
      float v0 = acc[j].x * inv, v1 = acc[j].y * inv;
      float v2 = acc[j].z * inv, v3 = acc[j].w * inv;
      u16 h0,l0,h1,l1,h2,l2,h3,l3;
      split_bf16(v0,h0,l0); split_bf16(v1,h1,l1);
      split_bf16(v2,h2,l2); split_bf16(v3,h3,l3);
      uint2 ph, pl;
      ph.x = (u32)h0 | ((u32)h1 << 16); ph.y = (u32)h2 | ((u32)h3 << 16);
      pl.x = (u32)l0 | ((u32)l1 << 16); pl.y = (u32)l2 | ((u32)l3 << 16);
      *(uint2*)(Ahi + r * LDA + h) = ph;
      *(uint2*)(Alo + r * LDA + h) = pl;
    }
  }
  __syncthreads();

  // ---- phase 2: 32x512 GEMM tile, bf16x3 WMMA ----
  // fragment layouts used here (C/D as in ISA 7.12.2; A/B K order is this code's own):
  //  A 16x32 bf16: lane holds 16 contiguous K of row lr (g selects K half), not the ISA table order
  //  B 32x16 bf16: lane holds 16 contiguous K of column lr (g selects K half), same order as A
  //  C/D f32:      vgpr r -> row r + 8*g, col lr
  const int g  = lane >> 4;
  const int lr = lane & 15;
  const int nbase = wv * 64;               // 8 waves x 64 cols = 512

  v8f acc[2][4] = {};                      // 2 M-tiles x 4 N-tiles

  for (int kk = 0; kk < 24; ++kk) {        // K = 768 = 24 * 32
    const int k0 = kk * 32;

    V16u ah[2], al[2];
    #pragma unroll
    for (int mt = 0; mt < 2; ++mt) {       // A fragments from LDS (4x ds_load_b128 each)
      const int e0 = (mt * 16 + lr) * LDA + k0 + g * 16;
      ah[mt].h[0] = *(const v8bf*)(Ahi + e0);
      ah[mt].h[1] = *(const v8bf*)(Ahi + e0 + 8);
      al[mt].h[0] = *(const v8bf*)(Alo + e0);
      al[mt].h[1] = *(const v8bf*)(Alo + e0 + 8);
    }

    #pragma unroll
    for (int nt = 0; nt < 4; ++nt) {
      const int e = nbase + nt * 16 + lr;
      const size_t wo = (size_t)e * H_ + k0 + g * 16;   // 32B-aligned, contiguous in K
      const v16bf bh = *(const v16bf*)(Whi + wo);
      const v16bf bl = *(const v16bf*)(Wlo + wo);
      #pragma unroll
      for (int mt = 0; mt < 2; ++mt) {     // reuse B across both M-tiles
        acc[mt][nt] = __builtin_amdgcn_wmma_f32_16x16x32_bf16(
            false, ah[mt].v, false, bh, (short)0, acc[mt][nt], false, false);
        acc[mt][nt] = __builtin_amdgcn_wmma_f32_16x16x32_bf16(
            false, ah[mt].v, false, bl, (short)0, acc[mt][nt], false, false);
        acc[mt][nt] = __builtin_amdgcn_wmma_f32_16x16x32_bf16(
            false, al[mt].v, false, bh, (short)0, acc[mt][nt], false, false);
      }
    }
  }

  // ---- epilogue: bias + store (C layout: row = mt*16 + g*8 + r, col = nbase+nt*16+lr) ----
  #pragma unroll
  for (int nt = 0; nt < 4; ++nt) {
    const int col = nbase + nt * 16 + lr;
    const float bv = bias[col];
    #pragma unroll
    for (int mt = 0; mt < 2; ++mt) {
      const size_t rbase = (size_t)(b * T_ + t0 + mt * 16 + g * 8);
      #pragma unroll
      for (int r2 = 0; r2 < 8; ++r2)
        out[(rbase + r2) * E_ + col] = acc[mt][nt][r2] + bv;
    }
  }
}

extern "C" void kernel_launch(void* const* d_in, const int* in_sizes, int n_in,
                              void* d_out, int out_size, void* d_ws, size_t ws_size,
                              hipStream_t stream) {
  const float* hs   = (const float*)d_in[0];   // [64,512,768]
  const int*   wid  = (const int*)  d_in[1];   // [64,512]
  const int*   tlen = (const int*)  d_in[2];   // [64]
  const float* W    = (const float*)d_in[3];   // [512,768]
  const float* bias = (const float*)d_in[4];   // [512]
  float*       out  = (float*)d_out;           // [64,256,512]

  u16* Whi = (u16*)d_ws;                       // 512*768 bf16
  u16* Wlo = Whi + (size_t)E_ * H_;            // 512*768 bf16  (total ws use: 1.5 MB)

  split_w_kernel<<<(E_ * H_ + 255) / 256, 256, 0, stream>>>(W, Whi, Wlo);

  const size_t smem = (size_t)(2 * ROWS * LDA) * sizeof(u16) + (ROWS + 1) * sizeof(int); // ~99.5 KB
  fused_pool_gemm<<<B_ * (T_ / ROWS), 256, smem, stream>>>(hs, wid, tlen, Whi, Wlo, bias, out);
}
